// Decoder_39402029974028
// MI455X (gfx1250) — matrix operand order repaired and checked by reading, not compiled, not run on hardware
//
#include <hip/hip_runtime.h>
#include <cstddef>

// ---- problem constants --------------------------------------------------
#define NB 64      // batch
#define NT 2048    // tokens
#define NS 64      // sentences
#define NH 256     // hidden
#define NE 128     // embed
#define NV 50000   // vocab
#define NX 50      // extra vocab
#define NH2 512    // 2H
#define NVX (NV + NX)

typedef __attribute__((ext_vector_type(16))) _Float16 v16h;
typedef __attribute__((ext_vector_type(8)))  float    v8f;

// ---- output offsets (tuple concatenated flat, f32) ----------------------
constexpr size_t OF_FINAL = 0;
constexpr size_t OF_H     = (size_t)NB * NVX;          // h_new  [B,H]
constexpr size_t OF_C     = OF_H    + (size_t)NB * NH; // c_new  [B,H]
constexpr size_t OF_CT    = OF_C    + (size_t)NB * NH; // c_t    [B,2H]
constexpr size_t OF_ATTN  = OF_CT   + (size_t)NB * NH2;// attn   [B,T]
constexpr size_t OF_COV   = OF_ATTN + (size_t)NB * NT; // coverage_next [B,T]
constexpr size_t OF_SCT   = OF_COV  + (size_t)NB * NT; // sent_c_t [B,2H]
constexpr size_t OF_SATTN = OF_SCT  + (size_t)NB * NH2;// sattn  [B,S]

// ---- workspace offsets (floats) -----------------------------------------
constexpr size_t WX      = 0;                          // x      [B,E]
constexpr size_t WGATES  = WX      + (size_t)NB * NE;  // gates  [B,4H]
constexpr size_t WSTH    = WGATES  + (size_t)NB * 4*NH;// s_t_hat[B,2H]
constexpr size_t WDEC    = WSTH    + (size_t)NB * NH2; // dec_fea[B,2H]
constexpr size_t WSDEC   = WDEC    + (size_t)NB * NH2; // sdec   [B,2H]
constexpr size_t WSCORES = WSDEC   + (size_t)NB * NH2; // scores [B,T]
constexpr size_t WSSC    = WSCORES + (size_t)NB * NT;  // sscores[B,S]
constexpr size_t WSEGSUM = WSSC    + (size_t)NB * NS;  // seg_sum[B,S]
constexpr size_t WSEGID  = WSEGSUM + (size_t)NB * NS;  // seg_id [B,T] (int)
constexpr size_t WPGEN   = WSEGID  + (size_t)NB * NT;  // p_gen  [B]
constexpr size_t WO1     = WPGEN   + (size_t)NB;       // o1 out [B,H]

__device__ __forceinline__ float sigf(float x) { return 1.0f / (1.0f + __expf(-x)); }

// ===== 1) x = concat(c_t_1, emb[y]) @ xc_W^T + xc_b =======================
__global__ void k_x(const int* __restrict__ y, const float* __restrict__ c_t_1,
                    const float* __restrict__ emb, const float* __restrict__ xcW,
                    const float* __restrict__ xcb, float* __restrict__ x) {
  int idx = blockIdx.x * blockDim.x + threadIdx.x;      // B*E
  if (idx >= NB * NE) return;
  int b = idx / NE, e = idx - b * NE;
  const float* w  = xcW + (size_t)e * (NH2 + NE);
  const float* ct = c_t_1 + (size_t)b * NH2;
  const float* em = emb + (size_t)y[b] * NE;
  float s = xcb[e];
  #pragma unroll 4
  for (int i = 0; i < NH2; ++i) s += ct[i] * w[i];
  #pragma unroll 4
  for (int i = 0; i < NE; ++i)  s += em[i] * w[NH2 + i];
  x[idx] = s;
}

// ===== 2) LSTM gates ======================================================
__global__ void k_gates(const float* __restrict__ x, const float* __restrict__ h0,
                        const float* __restrict__ Wih, const float* __restrict__ Whh,
                        const float* __restrict__ bih, const float* __restrict__ bhh,
                        float* __restrict__ gates) {
  int idx = blockIdx.x * blockDim.x + threadIdx.x;      // B*4H
  if (idx >= NB * 4 * NH) return;
  int b = idx / (4 * NH), j = idx - b * 4 * NH;
  const float* wi = Wih + (size_t)j * NE;
  const float* wh = Whh + (size_t)j * NH;
  const float* xb = x + (size_t)b * NE;
  const float* hb = h0 + (size_t)b * NH;
  float s = bih[j] + bhh[j];
  #pragma unroll 4
  for (int i = 0; i < NE; ++i) s += xb[i] * wi[i];
  #pragma unroll 4
  for (int i = 0; i < NH; ++i) s += hb[i] * wh[i];
  gates[idx] = s;
}

// ===== 3) LSTM cell -> h_new, c_new, s_t_hat ==============================
__global__ void k_cell(const float* __restrict__ gates, const float* __restrict__ c0,
                       float* __restrict__ out, float* __restrict__ sth) {
  int idx = blockIdx.x * blockDim.x + threadIdx.x;      // B*H
  if (idx >= NB * NH) return;
  int b = idx / NH, h = idx - b * NH;
  const float* g = gates + (size_t)b * 4 * NH;
  float ig = sigf(g[h]), fg = sigf(g[NH + h]);
  float gg = tanhf(g[2 * NH + h]), og = sigf(g[3 * NH + h]);
  float cn = fg * c0[idx] + ig * gg;
  float hn = og * tanhf(cn);
  out[OF_H + idx] = hn;
  out[OF_C + idx] = cn;
  sth[(size_t)b * NH2 + h]      = hn;
  sth[(size_t)b * NH2 + NH + h] = cn;
}

// ===== 4) dec_fea / sdec: s_t_hat @ W^T + b  (W: [2H,2H]) =================
__global__ void k_proj(const float* __restrict__ sth, const float* __restrict__ W,
                       const float* __restrict__ bias, float* __restrict__ o) {
  int idx = blockIdx.x * blockDim.x + threadIdx.x;      // B*2H
  if (idx >= NB * NH2) return;
  int b = idx / NH2, j = idx - b * NH2;
  const float* w = W + (size_t)j * NH2;
  const float* s = sth + (size_t)b * NH2;
  float acc = bias[j];
  #pragma unroll 4
  for (int i = 0; i < NH2; ++i) acc += s[i] * w[i];
  o[idx] = acc;
}

// ===== 5) attention scores: wave per (b,row) over 512 dims ================
__global__ void k_scores(const float* __restrict__ feat, const float* __restrict__ dec,
                         const float* __restrict__ covw, const float* __restrict__ cov,
                         const float* __restrict__ vw, float* __restrict__ o, int rows) {
  int lane = threadIdx.x & 31;
  int wave = blockIdx.x * (blockDim.x >> 5) + (threadIdx.x >> 5);
  if (wave >= NB * rows) return;
  int b = wave / rows;
  const float* f = feat + (size_t)wave * NH2;
  const float* d = dec + (size_t)b * NH2;
  float cv = cov ? cov[wave] : 0.0f;
  float s = 0.0f;
  int base = lane * 16;
  #pragma unroll
  for (int i = 0; i < 16; ++i) {
    int k = base + i;
    float e = f[k] + d[k];
    if (covw) e += cv * covw[k];
    s += tanhf(e) * vw[k];
  }
  #pragma unroll
  for (int off = 16; off; off >>= 1) s += __shfl_xor(s, off, 32);
  if (lane == 0) o[wave] = s;
}

// ===== 6) word softmax+mask+renorm, segment ids + segment sums ============
__global__ void k_word_norm(const float* __restrict__ sc, const float* __restrict__ mask,
                            const int* __restrict__ ends, float* __restrict__ attn,
                            float* __restrict__ seg_sum, int* __restrict__ seg_id) {
  __shared__ float red[256];
  __shared__ int   le[NS];
  __shared__ float ssum[NS];
  int b = blockIdx.x, tid = threadIdx.x;
  const float* s = sc + (size_t)b * NT;
  const float* m = mask + (size_t)b * NT;
  if (tid < NS) { le[tid] = ends[b * NS + tid]; ssum[tid] = 0.0f; }
  float mx = -3.4e38f;
  for (int t = tid; t < NT; t += 256) mx = fmaxf(mx, s[t]);
  red[tid] = mx; __syncthreads();
  for (int o = 128; o; o >>= 1) { if (tid < o) red[tid] = fmaxf(red[tid], red[tid + o]); __syncthreads(); }
  mx = red[0]; __syncthreads();
  float sum = 0.0f;
  for (int t = tid; t < NT; t += 256) {
    float z = __expf(s[t] - mx) * m[t];
    attn[(size_t)b * NT + t] = z;
    sum += z;
  }
  red[tid] = sum; __syncthreads();
  for (int o = 128; o; o >>= 1) { if (tid < o) red[tid] += red[tid + o]; __syncthreads(); }
  float inv = 1.0f / red[0]; __syncthreads();
  for (int t = tid; t < NT; t += 256) {
    float a = attn[(size_t)b * NT + t] * inv;
    attn[(size_t)b * NT + t] = a;
    int lo = 0, hi = NS;                       // searchsorted right: #(ends <= t)
    while (lo < hi) { int mid = (lo + hi) >> 1; if (le[mid] <= t) lo = mid + 1; else hi = mid; }
    seg_id[(size_t)b * NT + t] = lo;
    atomicAdd(&ssum[lo], a);
  }
  __syncthreads();
  if (tid < NS) seg_sum[b * NS + tid] = ssum[tid];
}

// ===== 7) sentence softmax+mask+renorm ====================================
__global__ void k_sent_norm(const float* __restrict__ sc, const float* __restrict__ mask,
                            float* __restrict__ sattn) {
  __shared__ float red[NS];
  int b = blockIdx.x, tid = threadIdx.x;     // blockDim = 64
  float v = sc[b * NS + tid];
  red[tid] = v; __syncthreads();
  for (int o = 32; o; o >>= 1) { if (tid < o) red[tid] = fmaxf(red[tid], red[tid + o]); __syncthreads(); }
  float mx = red[0]; __syncthreads();
  float z = __expf(v - mx) * mask[b * NS + tid];
  red[tid] = z; __syncthreads();
  for (int o = 32; o; o >>= 1) { if (tid < o) red[tid] += red[tid + o]; __syncthreads(); }
  sattn[b * NS + tid] = z / red[0];
}

// ===== 8) attn_sw + coverage_next =========================================
__global__ void k_cov(const float* __restrict__ attn, const int* __restrict__ seg_id,
                      const float* __restrict__ seg_sum, const float* __restrict__ sattn,
                      const float* __restrict__ coverage, float* __restrict__ cov_next) {
  int idx = blockIdx.x * blockDim.x + threadIdx.x;      // B*T
  if (idx >= NB * NT) return;
  int b = idx / NT;
  int sid = seg_id[idx];
  float asw = attn[idx] / seg_sum[b * NS + sid] * sattn[b * NS + sid];
  cov_next[idx] = coverage[idx] + asw;
}

// ===== 9) c_t = attn @ encoder_outputs (streaming) ========================
__global__ void k_ct(const float* __restrict__ attn, const float* __restrict__ enc,
                     float* __restrict__ ct) {
  __shared__ float la[128];
  int b = blockIdx.x, tid = threadIdx.x;     // blockDim = 512
  float acc = 0.0f;
  for (int t0 = 0; t0 < NT; t0 += 128) {
    if (tid < 128) la[tid] = attn[(size_t)b * NT + t0 + tid];
    __syncthreads();
    const float* e = enc + ((size_t)b * NT + t0) * NH2 + tid;
    #pragma unroll 4
    for (int j = 0; j < 128; ++j) acc += la[j] * e[(size_t)j * NH2];
    __syncthreads();
  }
  ct[(size_t)b * NH2 + tid] = acc;
}

__global__ void k_sct(const float* __restrict__ sattn, const float* __restrict__ senc,
                      float* __restrict__ sct) {
  __shared__ float la[NS];
  int b = blockIdx.x, tid = threadIdx.x;     // blockDim = 512
  if (tid < NS) la[tid] = sattn[b * NS + tid];
  __syncthreads();
  float acc = 0.0f;
  const float* e = senc + (size_t)b * NS * NH2 + tid;
  #pragma unroll
  for (int s = 0; s < NS; ++s) acc += la[s] * e[(size_t)s * NH2];
  sct[(size_t)b * NH2 + tid] = acc;
}

// ===== 10) p_gen ==========================================================
__global__ void k_pgen(const float* __restrict__ ct, const float* __restrict__ sth,
                       const float* __restrict__ x, const float* __restrict__ pgW,
                       const float* __restrict__ pgb, float* __restrict__ pg) {
  int b = threadIdx.x;                       // blockDim = 64
  float s = pgb[0];
  const float* c = ct + (size_t)b * NH2;
  const float* t = sth + (size_t)b * NH2;
  const float* xb = x + (size_t)b * NE;
  for (int i = 0; i < NH2; ++i) s += c[i] * pgW[i];
  for (int i = 0; i < NH2; ++i) s += t[i] * pgW[NH2 + i];
  for (int i = 0; i < NE; ++i)  s += xb[i] * pgW[2 * NH2 + i];
  pg[b] = sigf(s);
}

// ===== 11) o1: concat(h_new, c_t) @ o1_W^T + b ============================
__global__ void k_o1(const float* __restrict__ out, const float* __restrict__ ct,
                     const float* __restrict__ W, const float* __restrict__ bias,
                     float* __restrict__ o) {
  int idx = blockIdx.x * blockDim.x + threadIdx.x;      // B*H
  if (idx >= NB * NH) return;
  int b = idx / NH, j = idx - b * NH;
  const float* w = W + (size_t)j * 3 * NH;
  const float* h = out + OF_H + (size_t)b * NH;
  const float* c = ct + (size_t)b * NH2;
  float s = bias[j];
  #pragma unroll 4
  for (int i = 0; i < NH; ++i)  s += h[i] * w[i];
  #pragma unroll 4
  for (int i = 0; i < NH2; ++i) s += c[i] * w[NH + i];
  o[idx] = s;
}

// ===== 12) logits GEMM via WMMA f16 =======================================
// One wave per 16-column vocab tile; the wave carries ALL 64 batch rows as
// four 16x16 f32 accumulators, so o2_W is streamed from HBM exactly once
// (51 MB) and each B fragment feeds 4 WMMAs. 8 K-steps x 4 M-tiles = 32
// v_wmma_f32_16x16x32_f16 per wave.
__global__ void k_logits(const float* __restrict__ A /*[B,256]*/,
                         const float* __restrict__ W /*[V,256]*/,
                         const float* __restrict__ bias /*[V]*/,
                         float* __restrict__ D /*final region, stride NVX*/) {
  int n0 = blockIdx.x * 16;                  // vocab tile
  int lane = threadIdx.x;                    // blockDim = 32 (one wave)
  int half = lane >> 4;                      // 0 | 1
  int l16  = lane & 15;
  const float* brow = W + (size_t)(n0 + l16) * NH;          // K contiguous
  v8f acc[4] = {v8f{}, v8f{}, v8f{}, v8f{}};
  #pragma unroll
  for (int k0 = 0; k0 < NH; k0 += 32) {
    v16h bfr{};
    #pragma unroll
    for (int j = 0; j < 16; ++j)             // B 32x16: lane=N, 16 consec K per half
      bfr[j] = (_Float16)brow[k0 + half * 16 + j];
    #pragma unroll
    for (int mt = 0; mt < 4; ++mt) {         // batch tiles m0 = 0,16,32,48
      const float* arow = A + (size_t)(mt * 16 + l16) * NH; // K contiguous
      v16h a{};
      #pragma unroll
      for (int j = 0; j < 8; ++j) {          // A 16x32: 16 consec K per half (matches B)
        a[j]     = (_Float16)arow[k0 + half * 16 + j];      // K 0..7 | 16..23
        a[8 + j] = (_Float16)arow[k0 + half * 16 + 8 + j];  // K 8..15 | 24..31
      }
      acc[mt] = __builtin_amdgcn_wmma_f32_16x16x32_f16(false, a, false, bfr,
                                                       (short)0, acc[mt], false, false);
    }
  }
  float bv = bias[n0 + l16];
  #pragma unroll
  for (int mt = 0; mt < 4; ++mt) {
    #pragma unroll
    for (int r = 0; r < 8; ++r) {            // C/D: VGPR r -> M = r + 8*half
      int m = mt * 16 + r + half * 8;
      D[(size_t)m * NVX + n0 + l16] = acc[mt][r] + bv;
    }
  }
}

// ===== 13) in-place vocab softmax * p_gen, extras =========================
__global__ void k_final(float* __restrict__ fin, const float* __restrict__ pg,
                        const float* __restrict__ ez) {
  __shared__ float red[256];
  int b = blockIdx.x, tid = threadIdx.x;
  float* row = fin + (size_t)b * NVX;
  float mx = -3.4e38f;
  for (int v = tid; v < NV; v += 256) mx = fmaxf(mx, row[v]);
  red[tid] = mx; __syncthreads();
  for (int o = 128; o; o >>= 1) { if (tid < o) red[tid] = fmaxf(red[tid], red[tid + o]); __syncthreads(); }
  mx = red[0]; __syncthreads();
  float sum = 0.0f;
  for (int v = tid; v < NV; v += 256) sum += __expf(row[v] - mx);
  red[tid] = sum; __syncthreads();
  for (int o = 128; o; o >>= 1) { if (tid < o) red[tid] += red[tid + o]; __syncthreads(); }
  float scale = pg[b] / red[0]; __syncthreads();
  for (int v = tid; v < NV; v += 256) row[v] = __expf(row[v] - mx) * scale;
  if (tid < NX) row[NV + tid] = ez[b * NX + tid];
}

// ===== 14) scatter-add (1-p_gen)*attn at extended vocab ids ===============
__global__ void k_scatter(float* __restrict__ fin, const float* __restrict__ pg,
                          const float* __restrict__ attn, const int* __restrict__ evb) {
  int idx = blockIdx.x * blockDim.x + threadIdx.x;      // B*T
  if (idx >= NB * NT) return;
  int b = idx / NT;
  atomicAdd(&fin[(size_t)b * NVX + evb[idx]], (1.0f - pg[b]) * attn[idx]);
}

// =========================================================================
extern "C" void kernel_launch(void* const* d_in, const int* in_sizes, int n_in,
                              void* d_out, int out_size, void* d_ws, size_t ws_size,
                              hipStream_t stream) {
  const int*   y_t_1   = (const int*)  d_in[0];
  const float* h0      = (const float*)d_in[1];
  const float* c0      = (const float*)d_in[2];
  const int*   ends    = (const int*)  d_in[3];
  const float* enc_out = (const float*)d_in[4];
  const float* enc_ft  = (const float*)d_in[5];
  const float* enc_msk = (const float*)d_in[6];
  const float* senc_out= (const float*)d_in[7];
  const float* senc_ft = (const float*)d_in[8];
  const float* senc_msk= (const float*)d_in[9];
  const float* c_t_1   = (const float*)d_in[10];
  const float* ez      = (const float*)d_in[11];
  const int*   evb     = (const int*)  d_in[12];
  const float* cover   = (const float*)d_in[13];
  // d_in[14] = step (unused)
  const float* emb   = (const float*)d_in[15];
  const float* xcW   = (const float*)d_in[16];
  const float* xcb   = (const float*)d_in[17];
  const float* Wih   = (const float*)d_in[18];
  const float* Whh   = (const float*)d_in[19];
  const float* bih   = (const float*)d_in[20];
  const float* bhh   = (const float*)d_in[21];
  const float* dpW   = (const float*)d_in[22];
  const float* dpb   = (const float*)d_in[23];
  const float* vw    = (const float*)d_in[24];
  const float* wcw   = (const float*)d_in[25];
  const float* sdpW  = (const float*)d_in[26];
  const float* sdpb  = (const float*)d_in[27];
  const float* svw   = (const float*)d_in[28];
  const float* pgW   = (const float*)d_in[29];
  const float* pgb   = (const float*)d_in[30];
  const float* o1W   = (const float*)d_in[31];
  const float* o1b   = (const float*)d_in[32];
  const float* o2W   = (const float*)d_in[33];
  const float* o2b   = (const float*)d_in[34];

  float* out = (float*)d_out;
  float* ws  = (float*)d_ws;
  float* x      = ws + WX;
  float* gates  = ws + WGATES;
  float* sth    = ws + WSTH;
  float* dec    = ws + WDEC;
  float* sdec   = ws + WSDEC;
  float* scores = ws + WSCORES;
  float* sscor  = ws + WSSC;
  float* segsum = ws + WSEGSUM;
  int*   segid  = (int*)(ws + WSEGID);
  float* pg     = ws + WPGEN;
  float* o1o    = ws + WO1;

  k_x<<<(NB * NE + 255) / 256, 256, 0, stream>>>(y_t_1, c_t_1, emb, xcW, xcb, x);
  k_gates<<<(NB * 4 * NH + 255) / 256, 256, 0, stream>>>(x, h0, Wih, Whh, bih, bhh, gates);
  k_cell<<<(NB * NH + 255) / 256, 256, 0, stream>>>(gates, c0, out, sth);

  k_proj<<<(NB * NH2 + 255) / 256, 256, 0, stream>>>(sth, dpW, dpb, dec);
  k_proj<<<(NB * NH2 + 255) / 256, 256, 0, stream>>>(sth, sdpW, sdpb, sdec);

  // word scores: one wave per (b,t) -> B*T waves, 8 waves per 256-block
  k_scores<<<(NB * NT) / 8, 256, 0, stream>>>(enc_ft, dec, wcw, cover, vw, scores, NT);
  // sentence scores: B*S waves
  k_scores<<<(NB * NS) / 8, 256, 0, stream>>>(senc_ft, sdec, nullptr, nullptr, svw, sscor, NS);

  k_word_norm<<<NB, 256, 0, stream>>>(scores, enc_msk, ends, out + OF_ATTN, segsum, segid);
  k_sent_norm<<<NB, NS, 0, stream>>>(sscor, senc_msk, out + OF_SATTN);

  k_cov<<<(NB * NT + 255) / 256, 256, 0, stream>>>(out + OF_ATTN, segid, segsum,
                                                   out + OF_SATTN, cover, out + OF_COV);
  k_ct<<<NB, 512, 0, stream>>>(out + OF_ATTN, enc_out, out + OF_CT);
  k_sct<<<NB, 512, 0, stream>>>(out + OF_SATTN, senc_out, out + OF_SCT);

  k_pgen<<<1, NB, 0, stream>>>(out + OF_CT, sth, x, pgW, pgb, pg);
  k_o1<<<(NB * NH + 255) / 256, 256, 0, stream>>>(out, out + OF_CT, o1W, o1b, o1o);

  // WMMA logits GEMM: one wave per 16-wide vocab tile, all 64 batch rows
  k_logits<<<NV / 16, 32, 0, stream>>>(o1o, o2W, o2b, out + OF_FINAL);

  k_final<<<NB, 256, 0, stream>>>(out + OF_FINAL, pg, ez);
  k_scatter<<<(NB * NT + 255) / 256, 256, 0, stream>>>(out + OF_FINAL, pg,
                                                       out + OF_ATTN, evb);
}
